// NGCFProxy_42975442764396
// MI455X (gfx1250) — matrix operand order repaired and checked by reading, not compiled, not run on hardware
//
#include <hip/hip_runtime.h>

// CDNA5 / gfx1250: wave32, WMMA 16x16x32 f16 -> f32 accumulate.
typedef __attribute__((ext_vector_type(16))) _Float16 v16h;
typedef __attribute__((ext_vector_type(8)))  float    v8f;

#define NGCF_D 64

// ---------------------------------------------------------------------------
// Small utility kernels
// ---------------------------------------------------------------------------
__global__ void ngcf_init_ego(const float* __restrict__ ue, const float* __restrict__ ie,
                              float* __restrict__ ego, float* __restrict__ acc,
                              long uf_elems, long tot) {
  long t = (long)blockIdx.x * blockDim.x + threadIdx.x;
  if (t >= tot) return;
  float v = (t < uf_elems) ? ue[t] : ie[t - uf_elems];
  ego[t] = v;
  acc[t] = v;
}

__global__ void ngcf_degree(const int* __restrict__ er, const int* __restrict__ ec,
                            float* __restrict__ deg, int E, int UF) {
  int e = blockIdx.x * blockDim.x + threadIdx.x;
  if (e >= E) return;
  atomicAdd(&deg[er[e]], 1.0f);
  atomicAdd(&deg[UF + ec[e]], 1.0f);
}

__global__ void ngcf_dinv(const float* __restrict__ deg, float* __restrict__ dinv,
                          const int* __restrict__ pU, const int* __restrict__ pK,
                          int UF, int I, int n) {
  int r = blockIdx.x * blockDim.x + threadIdx.x;
  if (r >= n) return;
  int U = *pU;
  float kf = (float)(*pK);
  float base = deg[r];
  if (r >= UF)     base += kf * (float)(UF - U);  // item: + k * Fk
  else if (r >= U) base += kf * (float)I;          // fake user: + k * I
  dinv[r] = (base > 0.f) ? rsqrtf(base) : 0.f;     // isolated nodes never used
}

// Pack W1[0..L), W2[0..L) (row-major 64x64 f32) into WMMA B-fragment order, f16.
// dst layout: halfs[ m*4096 + (nt*2+kh)*512 + lane*16 + j ]
// B 32x16 f16 order used here (contiguous k, same as A): lane holds column N=lane&15; K = 32*kh + (lane>=16?16:0) + j
__global__ void ngcf_prepack(const float* __restrict__ W1, const float* __restrict__ W2,
                             _Float16* __restrict__ wpack, int L) {
  int tid = blockIdx.x * blockDim.x + threadIdx.x;
  int total = 2 * L * 64 * 64;
  if (tid >= total) return;
  int j    = tid & 15;
  int lane = (tid >> 4) & 31;
  int frag = (tid >> 9) & 7;   // nt*2 + kh
  int m    = tid >> 12;        // 0..L-1 -> W1[m]; L..2L-1 -> W2[m-L]
  int kh = frag & 1, nt = frag >> 1;
  const float* W = (m < L) ? (W1 + (size_t)m * 4096) : (W2 + (size_t)(m - L) * 4096);
  int K = 32 * kh + ((lane >= 16) ? 16 : 0) + j;
  int N = nt * 16 + (lane & 15);
  wpack[tid] = (_Float16)W[K * 64 + N];
}

// ---------------------------------------------------------------------------
// WMMA wave-level GEMM helper: 16x64 stripe, K=64, f16 in / f32 out.
// A 16x32 f16 order used here (contiguous k, same as B): lanes 0-15 hold row M=lane, K {0..15};
//                     lanes 16-31 hold row M=lane-16, K {16..31}.
// ---------------------------------------------------------------------------
__device__ __forceinline__ void ngcf_wave_wmma(const _Float16* __restrict__ lin,
                                               const _Float16* __restrict__ bfrag,
                                               int lane, v8f c[4]) {
  int row = lane & 15;
  int kb  = (lane >= 16) ? 16 : 0;
  v16h a0, a1;
#pragma unroll
  for (int j = 0; j < 8; ++j) {
    a0[j]     = lin[row * 64 + kb + j];
    a0[j + 8] = lin[row * 64 + kb + 8 + j];
    a1[j]     = lin[row * 64 + 32 + kb + j];
    a1[j + 8] = lin[row * 64 + 32 + kb + 8 + j];
  }
#pragma unroll
  for (int nt = 0; nt < 4; ++nt) {
    v8f t = {};
    v16h b0 = *(const v16h*)(bfrag + ((size_t)(nt * 2 + 0) * 32 + lane) * 16);
    v16h b1 = *(const v16h*)(bfrag + ((size_t)(nt * 2 + 1) * 32 + lane) * 16);
    t = __builtin_amdgcn_wmma_f32_16x16x32_f16(false, a0, false, b0, (short)0, t, false, false);
    t = __builtin_amdgcn_wmma_f32_16x16x32_f16(false, a1, false, b1, (short)0, t, false, false);
    c[nt] = t;
  }
}

// temp = ego @ W1[k]   (one wave per 16-row stripe; 4 waves per block)
__global__ __launch_bounds__(128) void ngcf_gemm(const float* __restrict__ X,
                                                 const _Float16* __restrict__ bfrag,
                                                 float* __restrict__ Out, int n) {
  __shared__ float lds[4][1024];
  int wave = threadIdx.x >> 5, lane = threadIdx.x & 31;
  int stripe = blockIdx.x * 4 + wave;
  int nStripes = (n + 15) >> 4;
  if (stripe >= nStripes) return;
  int r0 = stripe * 16;
  float* lw = lds[wave];
  _Float16* lin = (_Float16*)lw;
#pragma unroll
  for (int q = 0; q < 8; ++q) {           // stage 16x64 f32 tile -> f16 LDS
    int f = (lane * 8 + q) * 4;
    int row = r0 + (f >> 6);
    float4 v = make_float4(0.f, 0.f, 0.f, 0.f);
    if (row < n) v = *(const float4*)(X + (long)row * 64 + (f & 63));
    lin[f + 0] = (_Float16)v.x; lin[f + 1] = (_Float16)v.y;
    lin[f + 2] = (_Float16)v.z; lin[f + 3] = (_Float16)v.w;
  }
  v8f c[4];
  ngcf_wave_wmma(lin, bfrag, lane, c);
#pragma unroll
  for (int nt = 0; nt < 4; ++nt)          // C fragments -> LDS f32 tile
#pragma unroll
    for (int v = 0; v < 8; ++v) {
      int m = (lane < 16) ? v : (v + 8);
      lw[m * 64 + nt * 16 + (lane & 15)] = c[nt][v];
    }
#pragma unroll
  for (int q = 0; q < 8; ++q) {           // coalesced b128 stores
    int f = (lane * 8 + q) * 4;
    int row = r0 + (f >> 6);
    if (row < n) *(float4*)(Out + (long)row * 64 + (f & 63)) = *(const float4*)(lw + f);
  }
}

// Real-edge SpMM scatter for both operands (temp & ego), 64 threads per edge.
// Z[r] += w * dinv[c] * X[c]  for each directed edge; dinv[r] applied later.
__global__ void ngcf_scatter(const int* __restrict__ er, const int* __restrict__ ec,
                             const float* __restrict__ dinv,
                             const float* __restrict__ temp, const float* __restrict__ ego,
                             float* __restrict__ Z1, float* __restrict__ Z2, int E, int UF) {
  long t = (long)blockIdx.x * blockDim.x + threadIdx.x;
  if (t >= (long)E * 64) return;
  int e = (int)(t >> 6);
  int j = (int)(t & 63);
  int u = er[e];
  int c = UF + ec[e];
  float du = dinv[u], dc = dinv[c];
  long ub = (long)u * 64 + j, cb = (long)c * 64 + j;
  atomicAdd(&Z1[ub], dc * temp[cb]);
  atomicAdd(&Z1[cb], du * temp[ub]);
  atomicAdd(&Z2[ub], dc * ego[cb]);
  atomicAdd(&Z2[cb], du * ego[ub]);
}

// Dense fake block, reduced to column sums:
// sbuf[0:64)=sum_items dinv*temp, [64:128)=sum_items dinv*ego,
// [128:192)=sum_fake dinv*temp,  [192:256)=sum_fake dinv*ego.
__global__ void ngcf_fake_reduce(const float* __restrict__ temp, const float* __restrict__ ego,
                                 const float* __restrict__ dinv, float* __restrict__ sbuf,
                                 const int* __restrict__ pU, int UF, int I, int nbI, int ROWS) {
  int j = threadIdx.x;     // column
  int b = blockIdx.x;
  float sT = 0.f, sE = 0.f;
  if (b < nbI) {
    int r0 = b * ROWS, r1 = min(r0 + ROWS, I);
    for (int i = r0; i < r1; ++i) {
      long r = (long)(UF + i);
      float dv = dinv[r];
      sT += dv * temp[r * 64 + j];
      sE += dv * ego[r * 64 + j];
    }
    atomicAdd(&sbuf[j], sT);
    atomicAdd(&sbuf[64 + j], sE);
  } else {
    int U = *pU;
    int r0 = (b - nbI) * ROWS, r1 = min(r0 + ROWS, UF);
    for (int r = max(r0, U); r < r1; ++r) {   // only fake users [U, UF)
      float dv = dinv[r];
      sT += dv * temp[(long)r * 64 + j];
      sE += dv * ego[(long)r * 64 + j];
    }
    atomicAdd(&sbuf[128 + j], sT);
    atomicAdd(&sbuf[192 + j], sE);
  }
}

// Add dense fake-block contributions (runs after scatter; no concurrency).
__global__ void ngcf_apply_fake(float* __restrict__ Z1, float* __restrict__ Z2,
                                const float* __restrict__ sbuf,
                                const int* __restrict__ pU, const int* __restrict__ pK,
                                int UF) {
  long r = blockIdx.x;
  int j = threadIdx.x;
  int U = *pU;
  float kf = (float)(*pK);
  if (r < U) return;
  if (r < UF) { Z1[r * 64 + j] += kf * sbuf[j];       Z2[r * 64 + j] += kf * sbuf[64 + j]; }
  else        { Z1[r * 64 + j] += kf * sbuf[128 + j]; Z2[r * 64 + j] += kf * sbuf[192 + j]; }
}

// ego_new = leaky_relu(dinv*Z1 + temp + ((dinv*Z2) .* ego) @ W2); acc += ego_new
__global__ __launch_bounds__(128) void ngcf_finalize(
    float* __restrict__ ego, const float* __restrict__ temp,
    const float* __restrict__ Z1, const float* __restrict__ Z2,
    const float* __restrict__ dinv, const _Float16* __restrict__ bfrag,
    float* __restrict__ acc, int n) {
  __shared__ float lds[4][1024];
  int wave = threadIdx.x >> 5, lane = threadIdx.x & 31;
  int stripe = blockIdx.x * 4 + wave;
  int nStripes = (n + 15) >> 4;
  if (stripe >= nStripes) return;
  int r0 = stripe * 16;
  float* lw = lds[wave];
  _Float16* lin = (_Float16*)lw;
#pragma unroll
  for (int q = 0; q < 8; ++q) {           // stage A' = dinv .* Z2 .* ego (f16)
    int f = (lane * 8 + q) * 4;
    int row = r0 + (f >> 6);
    float4 z = make_float4(0.f, 0.f, 0.f, 0.f), e = z;
    float dv = 0.f;
    if (row < n) {
      z  = *(const float4*)(Z2 + (long)row * 64 + (f & 63));
      e  = *(const float4*)(ego + (long)row * 64 + (f & 63));
      dv = dinv[row];
    }
    lin[f + 0] = (_Float16)(dv * z.x * e.x);
    lin[f + 1] = (_Float16)(dv * z.y * e.y);
    lin[f + 2] = (_Float16)(dv * z.z * e.z);
    lin[f + 3] = (_Float16)(dv * z.w * e.w);
  }
  v8f c[4];
  ngcf_wave_wmma(lin, bfrag, lane, c);
#pragma unroll
  for (int nt = 0; nt < 4; ++nt)
#pragma unroll
    for (int v = 0; v < 8; ++v) {
      int m = (lane < 16) ? v : (v + 8);
      lw[m * 64 + nt * 16 + (lane & 15)] = c[nt][v];
    }
#pragma unroll
  for (int q = 0; q < 8; ++q) {           // fuse: + dinv*Z1 + temp, leaky, acc
    int f = (lane * 8 + q) * 4;
    int row = r0 + (f >> 6);
    if (row >= n) continue;
    long g = (long)row * 64 + (f & 63);
    float dv = dinv[row];
    float4 z1 = *(const float4*)(Z1 + g);
    float4 tp = *(const float4*)(temp + g);
    float4 bi = *(const float4*)(lw + f);
    float4 av = *(const float4*)(acc + g);
    float4 o;
    float v0 = dv * z1.x + tp.x + bi.x; o.x = (v0 > 0.f) ? v0 : 0.01f * v0;
    float v1 = dv * z1.y + tp.y + bi.y; o.y = (v1 > 0.f) ? v1 : 0.01f * v1;
    float v2 = dv * z1.z + tp.z + bi.z; o.z = (v2 > 0.f) ? v2 : 0.01f * v2;
    float v3 = dv * z1.w + tp.w + bi.w; o.w = (v3 > 0.f) ? v3 : 0.01f * v3;
    av.x += o.x; av.y += o.y; av.z += o.z; av.w += o.w;
    *(float4*)(ego + g) = o;
    *(float4*)(acc + g) = av;
  }
}

__global__ void ngcf_output(const float* __restrict__ acc, float* __restrict__ out,
                            float s, long tot) {
  long t = (long)blockIdx.x * blockDim.x + threadIdx.x;
  if (t >= tot) return;
  out[t] = acc[t] * s;
}

// ---------------------------------------------------------------------------
extern "C" void kernel_launch(void* const* d_in, const int* in_sizes, int n_in,
                              void* d_out, int out_size, void* d_ws, size_t ws_size,
                              hipStream_t stream) {
  const float* user_emb = (const float*)d_in[0];
  const float* item_emb = (const float*)d_in[1];
  const float* W1       = (const float*)d_in[2];
  const float* W2       = (const float*)d_in[3];
  // d_in[4..7]: MLP params — dead code (output multiplied by 0.0 in reference)
  const int* edge_row = (const int*)d_in[8];
  const int* edge_col = (const int*)d_in[9];
  const int* pU = (const int*)d_in[10];  // user_num (device scalar)
  const int* pK = (const int*)d_in[11];  // malicious_k (device scalar)

  const int d  = NGCF_D;
  const int UF = in_sizes[0] / d;
  const int I  = in_sizes[1] / d;
  const int E  = in_sizes[8];
  const int L  = in_sizes[2] / (d * d);
  const int n  = UF + I;
  const long nd = (long)n * d;

  // Workspace carve (256B aligned): 5 node-major f32 matrices + small arrays.
  char* ws = (char*)d_ws;
  size_t off = 0;
  auto carve = [&](size_t bytes) -> void* {
    void* p = ws + off;
    off += (bytes + 255) & ~(size_t)255;
    return p;
  };
  float* ego  = (float*)carve(nd * 4);
  float* acc  = (float*)carve(nd * 4);
  float* temp = (float*)carve(nd * 4);
  float* Z1   = (float*)carve(nd * 4);
  float* Z2   = (float*)carve(nd * 4);
  float* deg  = (float*)carve((size_t)n * 4);
  float* dinv = (float*)carve((size_t)n * 4);
  float* sbuf = (float*)carve(256 * 4);
  _Float16* wpack = (_Float16*)carve((size_t)2 * L * d * d * 2);
  if (off > ws_size) return;  // deterministic bail-out if scratch too small

  // Degree / normalization (structure only, shared by both layers).
  hipMemsetAsync(deg, 0, (size_t)n * 4, stream);
  ngcf_degree<<<(E + 255) / 256, 256, 0, stream>>>(edge_row, edge_col, deg, E, UF);
  ngcf_dinv<<<(n + 255) / 256, 256, 0, stream>>>(deg, dinv, pU, pK, UF, I, n);

  // Pre-pack W1/W2 into WMMA B-fragment order (f16).
  int totalW = 2 * L * d * d;
  ngcf_prepack<<<(totalW + 255) / 256, 256, 0, stream>>>(W1, W2, wpack, L);

  ngcf_init_ego<<<(int)((nd + 255) / 256), 256, 0, stream>>>(
      user_emb, item_emb, ego, acc, (long)UF * d, nd);

  const int nStripes = (n + 15) >> 4;
  const int gemmBlocks = (nStripes + 3) / 4;
  const int scatterBlocks = (int)(((long)E * 64 + 255) / 256);
  const int ROWS = 128;
  const int nbI = (I + ROWS - 1) / ROWS;
  const int nbU = (UF + ROWS - 1) / ROWS;

  for (int k = 0; k < L; ++k) {
    ngcf_gemm<<<gemmBlocks, 128, 0, stream>>>(ego, wpack + (size_t)k * d * d, temp, n);
    hipMemsetAsync(Z1, 0, nd * 4, stream);
    hipMemsetAsync(Z2, 0, nd * 4, stream);
    hipMemsetAsync(sbuf, 0, 256 * 4, stream);
    ngcf_scatter<<<scatterBlocks, 256, 0, stream>>>(edge_row, edge_col, dinv, temp, ego,
                                                    Z1, Z2, E, UF);
    ngcf_fake_reduce<<<nbI + nbU, 64, 0, stream>>>(temp, ego, dinv, sbuf, pU, UF, I, nbI, ROWS);
    ngcf_apply_fake<<<n, 64, 0, stream>>>(Z1, Z2, sbuf, pU, pK, UF);
    ngcf_finalize<<<gemmBlocks, 128, 0, stream>>>(ego, temp, Z1, Z2, dinv,
                                                  wpack + (size_t)(L + k) * d * d, acc, n);
  }
  ngcf_output<<<(int)((nd + 255) / 256), 256, 0, stream>>>(acc, (float*)d_out,
                                                           1.0f / (float)(L + 1), nd);
}
